// AFTFull_214748365537
// MI455X (gfx1250) — matrix operand order repaired and checked by reading, not compiled, not run on hardware
//
#include <hip/hip_runtime.h>
#include <hip/hip_bf16.h>

typedef _Float16 f16;
typedef __attribute__((ext_vector_type(16))) _Float16 v16h;
typedef __attribute__((ext_vector_type(8)))  _Float16 v8h;
typedef __attribute__((ext_vector_type(8)))  float    v8f;
typedef __attribute__((ext_vector_type(4)))  unsigned int u32x4;
typedef __attribute__((ext_vector_type(8)))  int          i32x8;
typedef __attribute__((ext_vector_type(4)))  int          i32x4;

#define B_DIM 8
#define QT    4096
#define KVT   4096
#define CDIM  512
#define HID   128
#define AKP   72   // padded LDS row stride (f16): 144B = 9*16B (matches TDM pad cfg)

static __device__ __forceinline__ v8f wmma_f16(v16h a, v16h b, v8f c) {
  // (neg_a, A, neg_b, B, c_mod, C, reuse_a, reuse_b)
  return __builtin_amdgcn_wmma_f32_16x16x32_f16(false, a, false, b, (short)0, c, false, false);
}

// --- TDM: async 2D tile load ew[row0..row0+127][k0..k0+63] -> LDS -----------
// D# group0/group1 per CDNA5 ISA 8.3/8.4. data_size=2B; tile 64x128;
// tensor 4096x4096, dim0 stride 4096; pad 4 DWORDs after every 32 DWORDs
// stored -> LDS row pitch 144B = AKP f16 elements.
static __device__ __forceinline__ void tdm_load_a_tile(const f16* gsrc,
                                                       unsigned int lds_byte_off) {
  unsigned long long ga = (unsigned long long)(size_t)gsrc;
  u32x4 g0;
  g0[0] = 1u;                                   // count=1 (valid user descriptor)
  g0[1] = lds_byte_off;                         // lds_addr (bytes)
  g0[2] = (unsigned int)ga;                     // global_addr[31:0]
  g0[3] = (unsigned int)((ga >> 32) & 0x1FFFFFFull) | (2u << 30);  // addr[56:32], type=2
  i32x8 g1;
  g1[0] = (1 << 16) | (1 << 20) | (4 << 22) | (3 << 25);  // data_size=2B, pad_en,
                                                          // pad_interval=32dw, pad=4dw
  g1[1] = (int)(4096u << 16);   // [47:32] atomic_barrier=0; [63:48] tensor_dim0.lo=4096
  g1[2] = (int)(4096u << 16);   // [79:64] tensor_dim0.hi=0; [95:80] tensor_dim1.lo=4096
  g1[3] = (int)(64u << 16);     // [111:96] tensor_dim1.hi=0; [127:112] tile_dim0=64
  g1[4] = 128;                  // [143:128] tile_dim1=128; [159:144] tile_dim2=0
  g1[5] = 4096;                 // [191:160] tensor_dim0_stride=4096 (low 32)
  g1[6] = 0;                    // stride.hi=0; tensor_dim1_stride.lo=0
  g1[7] = 0;
  i32x4 z4 = {0, 0, 0, 0};                      // groups 2/3 unused (2D tensor)
  i32x8 z8 = {0, 0, 0, 0, 0, 0, 0, 0};
  __builtin_amdgcn_tensor_load_to_lds(g0, g1, z4, z4, z8, 0);
}

// --- Fragment loaders -------------------------------------------------------
// A fragment 16x32 f16 (LDS or global), rows r0..r0+15, row stride ld, k-slab
// [k0,k0+32). Lanes 0-15: K=[0..8)+[16..24); lanes 16-31: K=[8..16)+[24..32).
static __device__ __forceinline__ v16h load_a_any(const f16* A, int ld, int r0,
                                                  int k0, int l16, int half) {
  const f16* p = A + (size_t)(r0 + l16) * ld + k0 + half * 8;
  v8h lo = *(const v8h*)p;
  v8h hi = *(const v8h*)(p + 16);
  v16h a;
#pragma unroll
  for (int i = 0; i < 8; ++i) { a[i] = lo[i]; a[i + 8] = hi[i]; }
  return a;
}

// A fragment from f32 row-major source (convert to f16 on the fly).
static __device__ __forceinline__ v16h load_a_f32(const float* A, int ld, int m0,
                                                  int k0, int l16, int half) {
  const float* p = A + (size_t)(m0 + l16) * ld + k0 + half * 8;
  v8f lo = *(const v8f*)p;
  v8f hi = *(const v8f*)(p + 16);
  v16h a;
#pragma unroll
  for (int i = 0; i < 8; ++i) { a[i] = (f16)lo[i]; a[i + 8] = (f16)hi[i]; }
  return a;
}

// B fragment 32x16 f16 from K-contiguous storage BT[n][k] (leading dim ldk).
static __device__ __forceinline__ v16h load_bt_f16(const f16* BT, int ldk, int n0,
                                                   int k0, int l16, int half) {
  const f16* p = BT + (size_t)(n0 + l16) * ldk + k0 + half * 8;
  v8h lo = *(const v8h*)p;
  v8h hi = *(const v8h*)(p + 16);
  v16h b;
#pragma unroll
  for (int i = 0; i < 8; ++i) { b[i] = lo[i]; b[i + 8] = hi[i]; }
  return b;
}

static __device__ __forceinline__ v16h load_bt_f32(const float* BT, int ldk, int n0,
                                                   int k0, int l16, int half) {
  const float* p = BT + (size_t)(n0 + l16) * ldk + k0 + half * 8;
  v8f lo = *(const v8f*)p;
  v8f hi = *(const v8f*)(p + 16);
  v16h b;
#pragma unroll
  for (int i = 0; i < 8; ++i) { b[i] = (f16)lo[i]; b[i + 8] = (f16)hi[i]; }
  return b;
}

// --- Kernel 1: ew = f16(exp(w_bias)), 4096x4096 -----------------------------
__global__ __launch_bounds__(256) void ew_exp_kernel(const float* __restrict__ w,
                                                     f16* __restrict__ ew) {
  size_t i = ((size_t)blockIdx.x * 256 + threadIdx.x) * 8;
  v8f x = *(const v8f*)(w + i);
  v8h y;
#pragma unroll
  for (int r = 0; r < 8; ++r) y[r] = (f16)__expf(x[r]);
  *(v8h*)(ew + i) = y;
}

// --- Kernel 2: qs = sigmoid(q @ qW^T + qW_b), f32 [B*QT, HID] ---------------
__global__ __launch_bounds__(256) void proj_q_kernel(const float* __restrict__ q,
                                                     const float* __restrict__ qw,
                                                     const float* __restrict__ qb,
                                                     float* __restrict__ qs) {
  const int tid = threadIdx.x, lane = tid & 31, wave = tid >> 5;
  const int l16 = lane & 15, half = lane >> 4;
  const int mw = wave & 3, hw = wave >> 2;
  const int mBase = blockIdx.x * 128 + mw * 32;
  const int hBase = hw * 64;

  v8f acc[2][4] = {};
  for (int kb = 0; kb < CDIM; kb += 32) {
    v16h a0 = load_a_f32(q, CDIM, mBase, kb, l16, half);
    v16h a1 = load_a_f32(q, CDIM, mBase + 16, kb, l16, half);
#pragma unroll
    for (int g = 0; g < 4; ++g) {
      v16h bf = load_bt_f32(qw, CDIM, hBase + g * 16, kb, l16, half);
      acc[0][g] = wmma_f16(a0, bf, acc[0][g]);
      acc[1][g] = wmma_f16(a1, bf, acc[1][g]);
    }
  }
#pragma unroll
  for (int rb = 0; rb < 2; ++rb)
#pragma unroll
    for (int g = 0; g < 4; ++g) {
      const int h = hBase + g * 16 + l16;
      const float bias = qb[h];
#pragma unroll
      for (int r = 0; r < 8; ++r) {
        const int m = mBase + rb * 16 + half * 8 + r;
        const float x = acc[rb][g][r] + bias;
        qs[(size_t)m * HID + h] = 1.0f / (1.0f + __expf(-x));
      }
    }
}

// --- Kernel 3: ekvT[b][n][j] f16, n in [0,256): [0,128)=exp(k)*v, [128,256)=exp(k)
__global__ __launch_bounds__(256) void proj_kv_kernel(const float* __restrict__ kv,
                                                      const float* __restrict__ kw,
                                                      const float* __restrict__ kbias,
                                                      const float* __restrict__ vw,
                                                      const float* __restrict__ vbias,
                                                      f16* __restrict__ ekvT) {
  const int tid = threadIdx.x, lane = tid & 31, wave = tid >> 5;
  const int l16 = lane & 15, half = lane >> 4;
  const int mw = wave & 3, hw = wave >> 2;
  const int mBase = blockIdx.x * 128 + mw * 32;
  const int hBase = hw * 64;

  v8f acck[2][4] = {};
  v8f accv[2][4] = {};
  for (int kb = 0; kb < CDIM; kb += 32) {
    v16h a0 = load_a_f32(kv, CDIM, mBase, kb, l16, half);
    v16h a1 = load_a_f32(kv, CDIM, mBase + 16, kb, l16, half);
#pragma unroll
    for (int g = 0; g < 4; ++g) {
      v16h bk = load_bt_f32(kw, CDIM, hBase + g * 16, kb, l16, half);
      v16h bv = load_bt_f32(vw, CDIM, hBase + g * 16, kb, l16, half);
      acck[0][g] = wmma_f16(a0, bk, acck[0][g]);
      acck[1][g] = wmma_f16(a1, bk, acck[1][g]);
      accv[0][g] = wmma_f16(a0, bv, accv[0][g]);
      accv[1][g] = wmma_f16(a1, bv, accv[1][g]);
    }
  }
#pragma unroll
  for (int rb = 0; rb < 2; ++rb)
#pragma unroll
    for (int g = 0; g < 4; ++g) {
      const int h = hBase + g * 16 + l16;
      const float kb_ = kbias[h], vb_ = vbias[h];
      const int m0 = mBase + rb * 16 + half * 8;  // 8 consecutive rows
      v8h pnum, pden;
#pragma unroll
      for (int r = 0; r < 8; ++r) {
        const float ek = __expf(acck[rb][g][r] + kb_);
        const float vv = accv[rb][g][r] + vb_;
        pnum[r] = (f16)(ek * vv);
        pden[r] = (f16)ek;
      }
      const int b = m0 >> 12;        // 128-row tiles never cross batch
      const int j = m0 & 4095;
      f16* base = ekvT + (size_t)b * 256 * KVT;
      *(v8h*)(base + (size_t)h * KVT + j) = pnum;            // num columns
      *(v8h*)(base + (size_t)(HID + h) * KVT + j) = pden;    // den columns
    }
}

// --- Kernel 4: out[b,t,h] = qs * (ew @ ekv) / (ew @ ek) ---------------------
// Per batch [4096x4096]x[4096x256]. Block tile M=128, N=256; 8 waves =
// 2 M-waves x 4 h-waves; wave tile M=64 x 32 h-cols x {num,den} = 16 accums.
// A (ew) staged to LDS by TDM (double buffered 64-k stages); B direct global.
__global__ __launch_bounds__(256) void aft_main_kernel(const f16* __restrict__ ew,
                                                       const f16* __restrict__ ekvT,
                                                       const float* __restrict__ qs,
                                                       float* __restrict__ out) {
  __shared__ f16 As[2][128 * AKP];

  const int tid = threadIdx.x, lane = tid & 31, wave = tid >> 5;
  const int l16 = lane & 15, half = lane >> 4;
  const int mw = wave & 1, hw = wave >> 1;
  const int b = blockIdx.y;
  const int mBlock = blockIdx.x * 128;
  const int mOff = mw * 64;
  const int hBase = hw * 32;
  const f16* Bp = ekvT + (size_t)b * 256 * KVT;
  const f16* Atile = ew + (size_t)mBlock * KVT;

  // Prologue: wave 0 kicks off the stage-0 DMA.
  if (wave == 0)
    tdm_load_a_tile(Atile, (unsigned int)(size_t)&As[0][0]);

  v8f acc[4][2][2] = {};
  const int NSTAGE = KVT / 64;
  for (int it = 0; it < NSTAGE; ++it) {
    const int buf = it & 1;
    if (wave == 0)
      __builtin_amdgcn_s_wait_tensorcnt(0);  // current buffer's DMA complete
    __syncthreads();                          // publish tile; prior readers done
    if (wave == 0 && it + 1 < NSTAGE)
      tdm_load_a_tile(Atile + (it + 1) * 64,
                      (unsigned int)(size_t)&As[1 - buf][0]);
    __builtin_prefetch(Bp + (size_t)(hBase + l16) * KVT + (it + 1) * 64, 0, 3);
    __builtin_prefetch(Bp + (size_t)(HID + hBase + l16) * KVT + (it + 1) * 64, 0, 3);
#pragma unroll
    for (int sub = 0; sub < 2; ++sub) {
      const int kg = it * 64 + sub * 32;  // global k (for B)
      const int ks = sub * 32;            // k within stage (for A)
      v16h af[4];
#pragma unroll
      for (int rblk = 0; rblk < 4; ++rblk)
        af[rblk] = load_a_any(&As[buf][0], AKP, mOff + rblk * 16, ks, l16, half);
#pragma unroll
      for (int g = 0; g < 2; ++g) {
        v16h bn = load_bt_f16(Bp, KVT, hBase + g * 16, kg, l16, half);
        v16h bd = load_bt_f16(Bp, KVT, HID + hBase + g * 16, kg, l16, half);
#pragma unroll
        for (int rblk = 0; rblk < 4; ++rblk) {
          acc[rblk][g][0] = wmma_f16(af[rblk], bn, acc[rblk][g][0]);
          acc[rblk][g][1] = wmma_f16(af[rblk], bd, acc[rblk][g][1]);
        }
      }
    }
  }

#pragma unroll
  for (int rblk = 0; rblk < 4; ++rblk)
#pragma unroll
    for (int g = 0; g < 2; ++g) {
      const int h = hBase + g * 16 + l16;
#pragma unroll
      for (int r = 0; r < 8; ++r) {
        const int m = mBlock + mOff + rblk * 16 + half * 8 + r;
        const size_t idx = ((size_t)b * QT + m) * HID + h;
        out[idx] = qs[idx] * acc[rblk][g][0][r] / acc[rblk][g][1][r];
      }
    }
}

// --- Host launcher ----------------------------------------------------------
extern "C" void kernel_launch(void* const* d_in, const int* in_sizes, int n_in,
                              void* d_out, int out_size, void* d_ws, size_t ws_size,
                              hipStream_t stream) {
  (void)in_sizes; (void)n_in; (void)out_size; (void)ws_size;
  const float* q      = (const float*)d_in[0];
  const float* kv     = (const float*)d_in[1];
  const float* qW_w   = (const float*)d_in[2];
  const float* qW_b   = (const float*)d_in[3];
  const float* kW_w   = (const float*)d_in[4];
  const float* kW_b   = (const float*)d_in[5];
  const float* vW_w   = (const float*)d_in[6];
  const float* vW_b   = (const float*)d_in[7];
  const float* w_bias = (const float*)d_in[8];
  float* out = (float*)d_out;

  char* ws = (char*)d_ws;
  f16*   ew   = (f16*)ws;                                          // 32 MB
  f16*   ekvT = (f16*)(ws + (size_t)QT * KVT * sizeof(f16));       // 16 MB
  float* qs   = (float*)(ws + (size_t)QT * KVT * sizeof(f16)
                            + (size_t)B_DIM * 2 * HID * KVT * sizeof(f16)); // 16 MB

  ew_exp_kernel<<<(QT * KVT) / (256 * 8), 256, 0, stream>>>(w_bias, ew);
  proj_q_kernel<<<(B_DIM * QT) / 128, 256, 0, stream>>>(q, qW_w, qW_b, qs);
  proj_kv_kernel<<<(B_DIM * QT) / 128, 256, 0, stream>>>(kv, kW_w, kW_b, vW_w, vW_b, ekvT);
  aft_main_kernel<<<dim3(QT / 128, B_DIM), 256, 0, stream>>>(ew, ekvT, qs, out);
}
